// BipartiteGCN_55825984913941
// MI455X (gfx1250) — compile-verified, matrix operand order checked by reading, not run on hardware
//
#include <hip/hip_runtime.h>
#include <hip/hip_bf16.h>

// ---------------------------------------------------------------------------
// MI455X (gfx1250) bipartite GCN.
// All d=128 GEMMs use v_wmma_f32_16x16x32_bf16 (bf16 in, f32 acc).
// GEMM waves now compute 2 M-tiles (32 rows) per wave with software-pipelined
// B-fragment LDS loads: each ds_load feeds two back-to-back WMMAs.
// Per-edge Wf matmul fused: gather+relu into LDS tiles, WMMA vs LDS-resident
// Wf, scatter-add via f32 global atomics (agg table fits in 192MB L2).
// ---------------------------------------------------------------------------

#define LN_EPS 1e-5f

typedef __attribute__((ext_vector_type(16))) __bf16 v16bf;
typedef __attribute__((ext_vector_type(8)))  __bf16 v8bf;
typedef __attribute__((ext_vector_type(8)))  float  v8f;

// Load a 16x32 bf16 WMMA operand fragment (A layout; B uses same layout on a
// pre-transposed [N][K] matrix). p = &mat[(lane&15)*ld + k0 + ((lane>>4)<<3)].
// ISA 7.12.2: per lane, elements 0..7 = K kh..kh+7, elements 8..15 = K kh+16..kh+23.
__device__ __forceinline__ v16bf load_frag(const __bf16* p) {
  v16bf a;
  *((v8bf*)&a)       = *((const v8bf*)p);
  *(((v8bf*)&a) + 1) = *((const v8bf*)(p + 16));
  return a;
}

// ---------------------------------------------------------------------------
// Weight prep: W[C][K][N] f32  ->  Wt[C][N][K] bf16 (transposed for B frags)
// ---------------------------------------------------------------------------
__global__ void transpose_convert(const float* __restrict__ W,
                                  __bf16* __restrict__ Wt,
                                  int K, int N, int total) {
  for (int idx = blockIdx.x * blockDim.x + threadIdx.x; idx < total;
       idx += gridDim.x * blockDim.x) {
    int c = idx / (K * N);
    int rem = idx - c * K * N;
    int k = rem / N;
    int n = rem - k * N;
    Wt[(size_t)c * K * N + (size_t)n * K + k] = (__bf16)W[idx];
  }
}

// ---------------------------------------------------------------------------
// Fourier positional encoding: x[N,2] -> [N,128] (per feature: 32 sin, 32 cos)
// ---------------------------------------------------------------------------
__global__ void fourier_kernel(const float* __restrict__ x, int N,
                               float* __restrict__ outF,
                               __bf16* __restrict__ outB) {
  int idx = blockIdx.x * blockDim.x + threadIdx.x;
  if (idx >= N * 128) return;
  int n = idx >> 7, c = idx & 127;
  int f = c >> 6, r = c & 63, l = r & 31;
  float freq = 3.14159265358979323846f * (float)(1u << l);
  float ang = x[n * 2 + f] * freq;
  float v = (r < 32) ? __sinf(ang) : __cosf(ang);
  outF[idx] = v;
  outB[idx] = (__bf16)v;
}

__global__ void zero_f32(float* __restrict__ p, size_t n) {
  for (size_t i = blockIdx.x * (size_t)blockDim.x + threadIdx.x; i < n;
       i += (size_t)gridDim.x * blockDim.x)
    p[i] = 0.f;
}

__global__ void scale_to_bf16(const float* __restrict__ in,
                              __bf16* __restrict__ out,
                              const float* __restrict__ scalePtr, size_t n) {
  float s = scalePtr[0];
  for (size_t i = blockIdx.x * (size_t)blockDim.x + threadIdx.x; i < n;
       i += (size_t)gridDim.x * blockDim.x)
    out[i] = (__bf16)(in[i] * s);
}

// ---------------------------------------------------------------------------
// out[M,128] = A[M,128](bf16) @ Wt^T + bias ; optional relu, f32/bf16 outputs.
// 256 threads = 8 waves; each wave does 2 M-tiles (32 rows) -> block = 256
// rows. Wt (32KB) staged in LDS; B frags pipelined, each feeds 2 WMMAs.
// ---------------------------------------------------------------------------
__global__ __launch_bounds__(256) void gemm_bf16_128(
    const __bf16* __restrict__ A, int M,
    const __bf16* __restrict__ Wt,    // [128][128] (n,k)
    const float* __restrict__ bias,   // [128] or null
    float* __restrict__ outF,         // may be null
    __bf16* __restrict__ outB,        // may be null
    int relu) {
  __shared__ __align__(16) __bf16 sW[128 * 128];
  const int tid = threadIdx.x;
  for (int i = tid; i < 128 * 16; i += 256) {
    int n = i >> 4, kc = i & 15;
    *(uint4*)(sW + n * 128 + kc * 8) = *(const uint4*)(Wt + (size_t)n * 128 + kc * 8);
  }

  const int wave = tid >> 5, lane = tid & 31;
  const int row0 = blockIdx.x * 256 + wave * 32;
  const int mrow = lane & 15;
  const int kh = (lane >> 4) << 3;
  int ar0 = row0 + mrow;       if (ar0 >= M) ar0 = M - 1;
  int ar1 = row0 + 16 + mrow;  if (ar1 >= M) ar1 = M - 1;

  // warm next block's A stream (global_prefetch_b8)
  if (row0 + 256 < M) __builtin_prefetch(A + (size_t)(row0 + 256) * 128, 0, 3);

  __syncthreads();

  v8f acc[2][8] = {};
#pragma unroll
  for (int ks = 0; ks < 4; ++ks) {
    v16bf a0 = load_frag(A + (size_t)ar0 * 128 + ks * 32 + kh);
    v16bf a1 = load_frag(A + (size_t)ar1 * 128 + ks * 32 + kh);
    v16bf b = load_frag(sW + mrow * 128 + ks * 32 + kh);
#pragma unroll
    for (int t = 0; t < 8; ++t) {
      v16bf bn = b;
      if (t < 7) bn = load_frag(sW + ((t + 1) * 16 + mrow) * 128 + ks * 32 + kh);
      acc[0][t] = __builtin_amdgcn_wmma_f32_16x16x32_bf16(
          false, a0, false, b, (short)0, acc[0][t], false, false);
      acc[1][t] = __builtin_amdgcn_wmma_f32_16x16x32_bf16(
          false, a1, false, b, (short)0, acc[1][t], false, false);
      b = bn;
    }
  }

  const int colb = lane & 15;
  const int rowoff = (lane >> 4) * 8;
#pragma unroll
  for (int u = 0; u < 2; ++u) {
#pragma unroll
    for (int t = 0; t < 8; ++t) {
      int col = t * 16 + colb;
      float bv = bias ? bias[col] : 0.f;
      for (int r = 0; r < 8; ++r) {
        int row = row0 + u * 16 + rowoff + r;
        if (row < M) {
          float v = acc[u][t][r] + bv;
          if (relu) v = v > 0.f ? v : 0.f;
          if (outF) outF[(size_t)row * 128 + col] = v;
          if (outB) outB[(size_t)row * 128 + col] = (__bf16)v;
        }
      }
    }
  }
}

// ---------------------------------------------------------------------------
// h1 = relu([A0, A1] @ Wo1 + bias) -> bf16.  K=256 in two 32KB LDS phases.
// Same 2-M-tile/wave blocking and pipelined B loads.
// ---------------------------------------------------------------------------
__global__ __launch_bounds__(256) void gemm_bf16_concat256(
    const __bf16* __restrict__ A0, const __bf16* __restrict__ A1, int M,
    const __bf16* __restrict__ Wt,    // [128][256] (n,k)
    const float* __restrict__ bias,
    __bf16* __restrict__ outB) {
  __shared__ __align__(16) __bf16 sW[128 * 128];
  const int tid = threadIdx.x;
  const int wave = tid >> 5, lane = tid & 31;
  const int row0 = blockIdx.x * 256 + wave * 32;
  const int mrow = lane & 15;
  const int kh = (lane >> 4) << 3;
  int ar0 = row0 + mrow;       if (ar0 >= M) ar0 = M - 1;
  int ar1 = row0 + 16 + mrow;  if (ar1 >= M) ar1 = M - 1;

  v8f acc[2][8] = {};
  for (int p = 0; p < 2; ++p) {
    __syncthreads();  // protect sW across phases
    for (int i = tid; i < 128 * 16; i += 256) {
      int n = i >> 4, kc = i & 15;
      *(uint4*)(sW + n * 128 + kc * 8) =
          *(const uint4*)(Wt + (size_t)n * 256 + p * 128 + kc * 8);
    }
    __syncthreads();
    const __bf16* A = p ? A1 : A0;
#pragma unroll
    for (int ks = 0; ks < 4; ++ks) {
      v16bf a0 = load_frag(A + (size_t)ar0 * 128 + ks * 32 + kh);
      v16bf a1 = load_frag(A + (size_t)ar1 * 128 + ks * 32 + kh);
      v16bf b = load_frag(sW + mrow * 128 + ks * 32 + kh);
#pragma unroll
      for (int t = 0; t < 8; ++t) {
        v16bf bn = b;
        if (t < 7) bn = load_frag(sW + ((t + 1) * 16 + mrow) * 128 + ks * 32 + kh);
        acc[0][t] = __builtin_amdgcn_wmma_f32_16x16x32_bf16(
            false, a0, false, b, (short)0, acc[0][t], false, false);
        acc[1][t] = __builtin_amdgcn_wmma_f32_16x16x32_bf16(
            false, a1, false, b, (short)0, acc[1][t], false, false);
        b = bn;
      }
    }
  }

  const int colb = lane & 15;
  const int rowoff = (lane >> 4) * 8;
#pragma unroll
  for (int u = 0; u < 2; ++u) {
#pragma unroll
    for (int t = 0; t < 8; ++t) {
      int col = t * 16 + colb;
      float bv = bias[col];
      for (int r = 0; r < 8; ++r) {
        int row = row0 + u * 16 + rowoff + r;
        if (row < M) {
          float v = acc[u][t][r] + bv;
          v = v > 0.f ? v : 0.f;
          outB[(size_t)row * 128 + col] = (__bf16)v;
        }
      }
    }
  }
}

// ---------------------------------------------------------------------------
// Fused edge kernel: per wave, 16 edges.
//   m = Ln[dst] + Rn[src] + e*We ; m = relu(m*sf) ; c = m @ Wf ; agg[dst] += c+bf
// 128 threads = 4 waves, 64 edges/block. Wf (32KB) + message tiles (16KB) in
// LDS. B-fragment LDS loads pipelined ahead of the consuming WMMA.
// ---------------------------------------------------------------------------
__global__ __launch_bounds__(128) void edge_message_scatter(
    int E,
    const long long* __restrict__ sIdx, const long long* __restrict__ dIdx,
    const float* __restrict__ eattr,
    const float* __restrict__ Ln,     // [Nr][128]
    const float* __restrict__ Rn,     // [Nl][128]
    const float* __restrict__ WeI,    // [128]
    const float* __restrict__ sfI,    // scalar
    const __bf16* __restrict__ WfT,   // [128][128] (n,k)
    const float* __restrict__ bfI,    // [128]
    float* __restrict__ agg) {        // [Nr][128]
  __shared__ __align__(16) __bf16 sWf[128 * 128];
  __shared__ __align__(16) __bf16 sM[4][16 * 128];
  const int tid = threadIdx.x;
  for (int i = tid; i < 128 * 16; i += 128) {
    int n = i >> 4, kc = i & 15;
    *(uint4*)(sWf + n * 128 + kc * 8) = *(const uint4*)(WfT + (size_t)n * 128 + kc * 8);
  }

  const int wave = tid >> 5, lane = tid & 31;
  const int e0 = blockIdx.x * 64 + wave * 16;
  const float sfv = sfI[0];

  for (int er = 0; er < 16; ++er) {
    int e = e0 + er;
    bool act = e < E;
    long long sI = 0, dI = 0;
    float eav = 0.f;
    if (act) { sI = sIdx[e]; dI = dIdx[e]; eav = eattr[e]; }
    for (int c = lane; c < 128; c += 32) {
      float v = 0.f;
      if (act) {
        v = Ln[(size_t)dI * 128 + c] + Rn[(size_t)sI * 128 + c] + eav * WeI[c];
        v *= sfv;
        v = v > 0.f ? v : 0.f;
      }
      sM[wave][er * 128 + c] = (__bf16)v;
    }
  }
  __syncthreads();

  const int mrow = lane & 15;
  const int kh = (lane >> 4) << 3;
  v8f acc[8] = {};
#pragma unroll
  for (int ks = 0; ks < 4; ++ks) {
    v16bf a = load_frag(&sM[wave][mrow * 128 + ks * 32 + kh]);
    v16bf b = load_frag(&sWf[mrow * 128 + ks * 32 + kh]);
#pragma unroll
    for (int t = 0; t < 8; ++t) {
      v16bf bn = b;
      if (t < 7) bn = load_frag(&sWf[((t + 1) * 16 + mrow) * 128 + ks * 32 + kh]);
      acc[t] = __builtin_amdgcn_wmma_f32_16x16x32_bf16(
          false, a, false, b, (short)0, acc[t], false, false);
      b = bn;
    }
  }

  const int colb = lane & 15;
  const int rowoff = (lane >> 4) * 8;
  float bv[8];
#pragma unroll
  for (int t = 0; t < 8; ++t) bv[t] = bfI[t * 16 + colb];
  for (int r = 0; r < 8; ++r) {
    int e = e0 + rowoff + r;
    if (e >= E) continue;
    long long dI = dIdx[e];
    float* row = agg + (size_t)dI * 128;
#pragma unroll
    for (int t = 0; t < 8; ++t)
      atomicAdd(row + t * 16 + colb, acc[t][r] + bv[t]);
  }
}

// ---------------------------------------------------------------------------
// LayerNorm over d=128, one wave per node. Writes the new feature tables
// (f32 + bf16 mirrors) and the JumpingKnowledge output slice.
// ---------------------------------------------------------------------------
__global__ __launch_bounds__(256) void layernorm_update(
    const float* __restrict__ H, int N,
    const float* __restrict__ g, const float* __restrict__ b,
    float* __restrict__ featF, __bf16* __restrict__ featB,
    float* __restrict__ out,    // side base pointer, node stride 256
    int layerOff) {
  int wave = threadIdx.x >> 5, lane = threadIdx.x & 31;
  int node = blockIdx.x * 8 + wave;
  if (node >= N) return;
  const float* h = H + (size_t)node * 128;
  float vals[4], s1 = 0.f, s2 = 0.f;
#pragma unroll
  for (int q = 0; q < 4; ++q) {
    float v = h[lane + 32 * q];
    vals[q] = v;
    s1 += v;
    s2 += v * v;
  }
#pragma unroll
  for (int off = 16; off > 0; off >>= 1) {
    s1 += __shfl_xor(s1, off, 32);
    s2 += __shfl_xor(s2, off, 32);
  }
  float mu = s1 * (1.f / 128.f);
  float var = s2 * (1.f / 128.f) - mu * mu;
  float inv = rsqrtf(var + LN_EPS);
#pragma unroll
  for (int q = 0; q < 4; ++q) {
    int c = lane + 32 * q;
    float o = (vals[q] - mu) * inv * g[c] + b[c];
    featF[(size_t)node * 128 + c] = o;
    featB[(size_t)node * 128 + c] = (__bf16)o;
    out[(size_t)node * 256 + layerOff + c] = o;
  }
}

// ---------------------------------------------------------------------------
// Orchestration
// ---------------------------------------------------------------------------
extern "C" void kernel_launch(void* const* d_in, const int* in_sizes, int n_in,
                              void* d_out, int out_size, void* d_ws, size_t ws_size,
                              hipStream_t stream) {
  (void)n_in; (void)out_size; (void)ws_size;
  const float* x_s   = (const float*)d_in[0];
  const float* x_t   = (const float*)d_in[1];
  const float* eattr = (const float*)d_in[2];
  const long long* eidx = (const long long*)d_in[3];   // int64 per reference
  const float* Wl  = (const float*)d_in[4];
  const float* bl  = (const float*)d_in[5];
  const float* We  = (const float*)d_in[6];
  const float* Wr  = (const float*)d_in[7];
  const float* sf  = (const float*)d_in[8];
  const float* Wf  = (const float*)d_in[9];
  const float* bfb = (const float*)d_in[10];
  const float* sp  = (const float*)d_in[11];
  const float* Wo1 = (const float*)d_in[12];
  const float* bo1 = (const float*)d_in[13];
  const float* Wo2 = (const float*)d_in[14];
  const float* bo2 = (const float*)d_in[15];
  const float* ln_g = (const float*)d_in[16];
  const float* ln_b = (const float*)d_in[17];

  const int NS = in_sizes[0] / 2;
  const int NT = in_sizes[1] / 2;
  const int E  = in_sizes[2];
  const long long* srcI = eidx;
  const long long* dstI = eidx + E;
  const int NMAX = NS > NT ? NS : NT;

  size_t off = 0;
  char* base = (char*)d_ws;
  auto carve = [&](size_t bytes) -> void* {
    void* p = base + off;
    off = (off + bytes + 255) & ~(size_t)255;
    return p;
  };
  float*  hsF  = (float*)carve((size_t)NS * 128 * 4);
  float*  htF  = (float*)carve((size_t)NT * 128 * 4);
  __bf16* hsB  = (__bf16*)carve((size_t)NS * 128 * 2);
  __bf16* htB  = (__bf16*)carve((size_t)NT * 128 * 2);
  float*  Ln   = (float*)carve((size_t)NMAX * 128 * 4);
  float*  Rn   = (float*)carve((size_t)NMAX * 128 * 4);
  float*  agg  = (float*)carve((size_t)NMAX * 128 * 4);
  __bf16* aggB = (__bf16*)carve((size_t)NMAX * 128 * 2);
  __bf16* h1B  = (__bf16*)carve((size_t)NMAX * 128 * 2);
  __bf16* WlT  = (__bf16*)carve((size_t)4 * 128 * 128 * 2);
  __bf16* WrT  = (__bf16*)carve((size_t)4 * 128 * 128 * 2);
  __bf16* WfT  = (__bf16*)carve((size_t)4 * 128 * 128 * 2);
  __bf16* Wo2T = (__bf16*)carve((size_t)4 * 128 * 128 * 2);
  __bf16* Wo1T = (__bf16*)carve((size_t)4 * 256 * 128 * 2);
  float*  h2   = Ln;  // Ln is dead after the edge kernel; reuse for h2

  // weight transposes (f32 -> bf16, [C][K][N] -> [C][N][K])
  {
    int tot = 4 * 128 * 128;
    int gb = (tot + 255) / 256;
    transpose_convert<<<gb, 256, 0, stream>>>(Wl,  WlT,  128, 128, tot);
    transpose_convert<<<gb, 256, 0, stream>>>(Wr,  WrT,  128, 128, tot);
    transpose_convert<<<gb, 256, 0, stream>>>(Wf,  WfT,  128, 128, tot);
    transpose_convert<<<gb, 256, 0, stream>>>(Wo2, Wo2T, 128, 128, tot);
    int tot1 = 4 * 256 * 128;
    transpose_convert<<<(tot1 + 255) / 256, 256, 0, stream>>>(Wo1, Wo1T, 256, 128, tot1);
  }
  fourier_kernel<<<(NS * 128 + 255) / 256, 256, 0, stream>>>(x_s, NS, hsF, hsB);
  fourier_kernel<<<(NT * 128 + 255) / 256, 256, 0, stream>>>(x_t, NT, htF, htB);

  float* out_s = (float*)d_out;
  float* out_t = out_s + (size_t)NS * 256;

  for (int l = 0; l < 2; ++l) {
    for (int hvs = 0; hvs < 2; ++hvs) {     // 0: s->t (updates ht), 1: t->s (updates hs)
      int i = 2 * l + hvs;
      const __bf16* leftB  = hvs ? htB : hsB;
      float*  rightF = hvs ? hsF : htF;
      __bf16* rightB = hvs ? hsB : htB;
      int Nl = hvs ? NT : NS;
      int Nr = hvs ? NS : NT;
      const long long* sI = hvs ? dstI : srcI;
      const long long* dI = hvs ? srcI : dstI;
      float* outSide = hvs ? out_s : out_t;

      int gbR = (Nr + 255) / 256;
      int gbL = (Nl + 255) / 256;
      // Ln = right @ Wl + bl ; Rn = left @ Wr
      gemm_bf16_128<<<gbR, 256, 0, stream>>>(rightB, Nr, WlT + (size_t)i * 16384,
                                             bl + i * 128, Ln, nullptr, 0);
      gemm_bf16_128<<<gbL, 256, 0, stream>>>(leftB, Nl, WrT + (size_t)i * 16384,
                                             nullptr, Rn, nullptr, 0);
      size_t aggN = (size_t)Nr * 128;
      zero_f32<<<1024, 256, 0, stream>>>(agg, aggN);
      edge_message_scatter<<<(E + 63) / 64, 128, 0, stream>>>(
          E, sI, dI, eattr, Ln, Rn, We + i * 128, sf + i,
          WfT + (size_t)i * 16384, bfb + i * 128, agg);
      scale_to_bf16<<<1024, 256, 0, stream>>>(agg, aggB, sp + i, aggN);
      // h1 = relu([agg*sp, right] @ Wo1 + bo1) ; h2 = h1 @ Wo2 + bo2
      gemm_bf16_concat256<<<gbR, 256, 0, stream>>>(aggB, rightB, Nr,
                                                   Wo1T + (size_t)i * 32768,
                                                   bo1 + i * 128, h1B);
      gemm_bf16_128<<<gbR, 256, 0, stream>>>(h1B, Nr, Wo2T + (size_t)i * 16384,
                                             bo2 + i * 128, h2, nullptr, 0);
      layernorm_update<<<(Nr + 7) / 8, 256, 0, stream>>>(
          h2, Nr, ln_g + i * 128, ln_b + i * 128, rightF, rightB,
          outSide, l * 128);
    }
  }
}
